// SignedGraphReasoning_81990925681012
// MI455X (gfx1250) — matrix operand order repaired and checked by reading, not compiled, not run on hardware
//
#include <hip/hip_runtime.h>

typedef _Float16 v16h __attribute__((ext_vector_type(16)));
typedef _Float16 h8   __attribute__((ext_vector_type(8)));
typedef float    v8f  __attribute__((ext_vector_type(8)));

#define NUM_NODES 6
#define IN_FEAT   196
#define KPAD      224      // 196 padded to 7 k-tiles of 32
#define OUT_FEAT  128
#define BATCH     32768
#define BB        8        // batches per workgroup
#define ROWS      48       // BB * 6 node rows = 3 M-tiles of 16
#define NT1       14       // N-tiles for attn@G (224/16)
#define NT2       8        // N-tiles for nm@W^T (128/16)
#define KT        7        // K-tiles (224/32)

#define GB_ELEMS (KT * NT1 * 32 * 16)   // 50176 halves, fragment-ordered G
#define WB_ELEMS (KT * NT2 * 32 * 16)   // 28672 halves, fragment-ordered W^T

// ---------------- prep: pack G and W^T into WMMA-B fragment order ----------
// B-matrix 32x16 f16, k order used by this code (wave32): lane -> N = lane&15; half h (0..15) ->
// K = (lane>>4)*16 + h. We store each fragment's 16 halves contiguously per
// lane so the main kernel does one 32B load per B fragment.
__global__ void sgr_prep(const float* __restrict__ G, const float* __restrict__ W,
                         _Float16* __restrict__ GB, _Float16* __restrict__ WB) {
  int t = blockIdx.x * blockDim.x + threadIdx.x;
  if (t < GB_ELEMS) {
    int h = t & 15, lane = (t >> 4) & 31;
    int frag = t >> 9;            // kt*NT1 + nt
    int nt = frag % NT1, kt = frag / NT1;
    int n = nt * 16 + (lane & 15);
    int k = kt * 32 + (lane >> 4) * 16 + h;
    float v = (k < IN_FEAT && n < IN_FEAT) ? G[k * IN_FEAT + n] : 0.f;
    GB[t] = (_Float16)v;
  }
  if (t < WB_ELEMS) {
    int h = t & 15, lane = (t >> 4) & 31;
    int frag = t >> 9;            // kt*NT2 + nt
    int nt = frag % NT2, kt = frag / NT2;
    int n = nt * 16 + (lane & 15);                 // output channel (0..127)
    int k = kt * 32 + (lane >> 4) * 16 + h;        // input feature
    float v = (k < IN_FEAT) ? W[n * IN_FEAT + k] : 0.f;   // W^T element
    WB[t] = (_Float16)v;
  }
}

// A-matrix 16x32 f16, k order used by this code: lane -> M = lane&15; half h: K = (lane>>4)*16 + h
//  => two contiguous 8-half chunks at +0 and +8 halves.
__device__ __forceinline__ v16h load_a_frag(const _Float16* rowbase, int kt, int hi) {
  const h8* p0 = (const h8*)(rowbase + kt * 32 + hi * 16);
  const h8* p1 = (const h8*)(rowbase + kt * 32 + hi * 16 + 8);
  h8 lo = *p0, hh = *p1;
  return __builtin_shufflevector(lo, hh, 0,1,2,3,4,5,6,7,8,9,10,11,12,13,14,15);
}

__device__ __forceinline__ v8f wmma_f16(v16h a, v16h b, v8f c) {
  return __builtin_amdgcn_wmma_f32_16x16x32_f16(false, a, false, b, (short)0, c,
                                                false, false);
}

__global__ __launch_bounds__(256)
void sgr_main(const float* __restrict__ x, const float* __restrict__ Pm,
              const float* __restrict__ bias,
              const _Float16* __restrict__ GB, const _Float16* __restrict__ WB,
              float* __restrict__ out) {
  __shared__ __align__(32) _Float16 sAttn[ROWS * KPAD];   // softmax attn, f16
  __shared__ __align__(32) _Float16 sTile[ROWS * KPAD];   // T, then node_mixed
  __shared__ __align__(16) float sS[ROWS * ROWS];         // raw S tiles (48x48)
  __shared__ float sScale[ROWS], sOff[ROWS], sA[BB * 36];

  const int tid  = threadIdx.x;
  const int wave = tid >> 5;
  const int lane = tid & 31;
  const int hi   = lane >> 4;
  const int c16  = lane & 15;
  const int bb0  = blockIdx.x * BB;

  float* outE = out;                                        // [B,6,128]
  float* outA = out + (size_t)BATCH * NUM_NODES * OUT_FEAT; // [B,6,6]

  // ---- Phase 1: per-row min/max, alignment affine, softmax into sAttn ----
  // aligned = off + sign*(x-min)/(max-min+1e-8).  max(5*aligned) == 5, so
  // attn = exp(5*aligned - 5)/Z with no explicit max pass.
  for (int r = wave; r < ROWS; r += 8) {
    int b = r / 6, n = r % 6;
    const float* xr = x + ((size_t)(bb0 + b) * 6 + n) * IN_FEAT;
    float mn = 3.0e38f, mx = -3.0e38f;
    for (int k = lane; k < IN_FEAT; k += 32) {
      float v = xr[k];
      mn = fminf(mn, v);
      mx = fmaxf(mx, v);
    }
    for (int m = 16; m; m >>= 1) {
      mn = fminf(mn, __shfl_xor(mn, m, 32));
      mx = fmaxf(mx, __shfl_xor(mx, m, 32));
    }
    float sign = (n & 1) ? -1.f : 1.f;
    float offv = (n & 1) ? 1.f : 0.f;
    float inv  = 1.f / (mx - mn + 1e-8f);
    float a_s  = sign * inv;
    float a_o  = offv - sign * mn * inv;
    if (lane == 0) { sScale[r] = a_s; sOff[r] = a_o; }
    float s = 0.f;
    for (int k = lane; k < KPAD; k += 32) {
      float e = 0.f;
      if (k < IN_FEAT) {
        float al = a_o + a_s * xr[k];
        e = __expf(5.f * al - 5.f);
        s += e;
      }
      sAttn[r * KPAD + k] = (_Float16)e;   // also zero-pads K 196..223
    }
    for (int m = 16; m; m >>= 1) s += __shfl_xor(s, m, 32);
    float invZ = 1.f / s;
    for (int k = lane; k < IN_FEAT; k += 32)
      sAttn[r * KPAD + k] = (_Float16)((float)sAttn[r * KPAD + k] * invZ);
  }
  __syncthreads();

  // ---- Phase 2: T = attn @ G via WMMA (3 M-tiles per N-tile) ----
  for (int nt = wave; nt < NT1; nt += 8) {
    v8f c0 = {0.f,0.f,0.f,0.f,0.f,0.f,0.f,0.f}, c1 = c0, c2 = c0;
    for (int kt = 0; kt < KT; ++kt) {
      v16h bf = *(const v16h*)(GB + (((kt * NT1 + nt) * 32 + lane) << 4));
      v16h a0 = load_a_frag(sAttn + ( 0 + c16) * KPAD, kt, hi);
      v16h a1 = load_a_frag(sAttn + (16 + c16) * KPAD, kt, hi);
      v16h a2 = load_a_frag(sAttn + (32 + c16) * KPAD, kt, hi);
      c0 = wmma_f16(a0, bf, c0);
      c1 = wmma_f16(a1, bf, c1);
      c2 = wmma_f16(a2, bf, c2);
    }
    int col = nt * 16 + c16;
    for (int r = 0; r < 8; ++r) {          // C/D layout: M = hi*8 + r
      sTile[( 0 + hi * 8 + r) * KPAD + col] = (_Float16)c0[r];
      sTile[(16 + hi * 8 + r) * KPAD + col] = (_Float16)c1[r];
      sTile[(32 + hi * 8 + r) * KPAD + col] = (_Float16)c2[r];
    }
  }
  __syncthreads();

  // ---- Phase 3a: S = T @ attn^T via WMMA (9 tiles of 16x16) ----
  // B-fragment of attn^T: element (K=p, N=j) = attn[j][p] -> per lane this is
  // 16 contiguous halves of an sAttn row at offset kt*32 + hi*16.
  for (int t = wave; t < 9; t += 8) {
    int mt = t / 3, jt = t % 3;
    v8f c = {0.f,0.f,0.f,0.f,0.f,0.f,0.f,0.f};
    const _Float16* brow = sAttn + (jt * 16 + c16) * KPAD + hi * 16;
    const _Float16* arow = sTile + (mt * 16 + c16) * KPAD;
    for (int kt = 0; kt < KT; ++kt) {
      v16h af = load_a_frag(arow, kt, hi);
      v16h bf = *(const v16h*)(brow + kt * 32);
      c = wmma_f16(af, bf, c);
    }
    int col = jt * 16 + c16;
    for (int r = 0; r < 8; ++r)
      sS[(mt * 16 + hi * 8 + r) * ROWS + col] = c[r];
  }
  __syncthreads();

  // ---- Phase 3b: A = P * (S + 0.1), block-diagonal 6x6 extraction ----
  for (int idx = tid; idx < BB * 36; idx += 256) {
    int b = idx / 36, rem = idx % 36;
    int i = rem / 6, j = rem % 6;
    float s = sS[(b * 6 + i) * ROWS + (b * 6 + j)];
    float a = Pm[rem] * (s + 0.1f);
    sA[idx] = a;
    outA[(size_t)(bb0 + b) * 36 + rem] = a;
  }
  __syncthreads();

  // ---- Phase 4: node_mixed = A @ aligned (recompute aligned from x) ----
  for (int t = tid; t < BB * KPAD; t += 256) {
    int b = t / KPAD, k = t % KPAD;
    if (k < IN_FEAT) {
      const float* xb = x + (size_t)(bb0 + b) * 6 * IN_FEAT;
      float al[6];
      for (int j = 0; j < 6; ++j) {
        int r = b * 6 + j;
        al[j] = sOff[r] + sScale[r] * xb[j * IN_FEAT + k];
      }
      for (int i = 0; i < 6; ++i) {
        float acc = 0.f;
        for (int j = 0; j < 6; ++j) acc += sA[b * 36 + i * 6 + j] * al[j];
        sTile[(b * 6 + i) * KPAD + k] = (_Float16)acc;
      }
    } else {
      for (int i = 0; i < 6; ++i)
        sTile[(b * 6 + i) * KPAD + k] = (_Float16)0.f;
    }
  }
  __syncthreads();

  // ---- Phase 5: out = leaky(nm @ W^T + bias) via WMMA, 1 N-tile/wave ----
  {
    int nt = wave;   // exactly 8 N-tiles
    v8f c0 = {0.f,0.f,0.f,0.f,0.f,0.f,0.f,0.f}, c1 = c0, c2 = c0;
    for (int kt = 0; kt < KT; ++kt) {
      v16h bf = *(const v16h*)(WB + (((kt * NT2 + nt) * 32 + lane) << 4));
      v16h a0 = load_a_frag(sTile + ( 0 + c16) * KPAD, kt, hi);
      v16h a1 = load_a_frag(sTile + (16 + c16) * KPAD, kt, hi);
      v16h a2 = load_a_frag(sTile + (32 + c16) * KPAD, kt, hi);
      c0 = wmma_f16(a0, bf, c0);
      c1 = wmma_f16(a1, bf, c1);
      c2 = wmma_f16(a2, bf, c2);
    }
    int col = nt * 16 + c16;
    float bv = bias[col];
    size_t rbase = (size_t)bb0 * 6;
    for (int r = 0; r < 8; ++r) {
      float v0 = c0[r] + bv; v0 = (v0 >= 0.f) ? v0 : 0.2f * v0;
      float v1 = c1[r] + bv; v1 = (v1 >= 0.f) ? v1 : 0.2f * v1;
      float v2 = c2[r] + bv; v2 = (v2 >= 0.f) ? v2 : 0.2f * v2;
      outE[(rbase +  0 + hi * 8 + r) * OUT_FEAT + col] = v0;
      outE[(rbase + 16 + hi * 8 + r) * OUT_FEAT + col] = v1;
      outE[(rbase + 32 + hi * 8 + r) * OUT_FEAT + col] = v2;
    }
  }
}

extern "C" void kernel_launch(void* const* d_in, const int* in_sizes, int n_in,
                              void* d_out, int out_size, void* d_ws, size_t ws_size,
                              hipStream_t stream) {
  (void)in_sizes; (void)n_in; (void)out_size; (void)ws_size;
  const float* x  = (const float*)d_in[0];   // [32768,6,14,14]
  const float* P  = (const float*)d_in[1];   // [6,6]
  const float* G  = (const float*)d_in[2];   // [196,196]
  const float* W  = (const float*)d_in[3];   // [128,196]
  const float* b  = (const float*)d_in[4];   // [128]

  _Float16* GB = (_Float16*)d_ws;            // 50176 halves
  _Float16* WB = GB + GB_ELEMS;              // 28672 halves (total ~158 KB)

  sgr_prep<<<dim3((GB_ELEMS + 255) / 256), dim3(256), 0, stream>>>(G, W, GB, WB);
  sgr_main<<<dim3(BATCH / BB), dim3(256), 0, stream>>>(
      x, P, b, GB, WB, (float*)d_out);
}
